// GroupedSTELinear_64665027609324
// MI455X (gfx1250) — matrix operand order repaired and checked by reading, not compiled, not run on hardware
//
#include <hip/hip_runtime.h>
#include <stdint.h>

// ---------------------------------------------------------------------------
// GroupedSTELinear forward for MI455X (gfx1250, wave32, WMMA, async-to-LDS).
// out = x @ (sign(wl) * softplus_scale).T  ==> bf16 GEMM, f32 accumulate.
// M=16384, N=2048, K=2048 (derived from in_sizes at launch).
// ---------------------------------------------------------------------------

typedef __attribute__((ext_vector_type(16))) __bf16 v16bf;
typedef __attribute__((ext_vector_type(8)))  float  v8f;
typedef __attribute__((ext_vector_type(4)))  int    v4i;

#define QGROUP 128
#define QEPS   1e-6f

#define BM 128
#define BN 128
#define BK 32
#define LDT 40   // padded LDS row stride in bf16 elems (80 B = 20 dwords -> conflict-free frag reads)

__device__ __forceinline__ unsigned short f2bf_rne(float f) {
  unsigned u = __builtin_bit_cast(unsigned, f);
  u += 0x7fffu + ((u >> 16) & 1u);   // round-to-nearest-even into bf16
  return (unsigned short)(u >> 16);
}

// --- CDNA5 async global->LDS (ASYNCcnt-tracked, no staging VGPRs) ----------
// Builtin signature (from clang diagnostic): (int4 AS1*, int4 AS3*, imm, imm).
typedef __attribute__((address_space(1))) v4i* as1_v4i;
typedef __attribute__((address_space(3))) v4i* as3_v4i;

__device__ __forceinline__ void async_copy_b128(const void* gsrc, void* ldst) {
#if __has_builtin(__builtin_amdgcn_global_load_async_to_lds_b128)
  as1_v4i g = (as1_v4i)(uintptr_t)gsrc;                  // inttoptr -> AS1
  as3_v4i l = (as3_v4i)(unsigned)(uintptr_t)ldst;        // low 32b = LDS offset
  __builtin_amdgcn_global_load_async_to_lds_b128(g, l, 0, 0);
#else
  unsigned lds32 = (unsigned)(uintptr_t)ldst;
  unsigned long long g64 = (unsigned long long)(uintptr_t)gsrc;
  asm volatile("global_load_async_to_lds_b128 %0, %1, off"
               :: "v"(lds32), "v"(g64) : "memory");
#endif
}

__device__ __forceinline__ void wait_async0() {
#if __has_builtin(__builtin_amdgcn_s_wait_asynccnt)
  __builtin_amdgcn_s_wait_asynccnt(0);
#else
  asm volatile("s_wait_asynccnt 0x0" ::: "memory");
#endif
}

// --- Kernel 1: dequantize sign weights to bf16 ------------------------------
__global__ void dequant_w_kernel(const float* __restrict__ wl,
                                 const float* __restrict__ rs,
                                 unsigned short* __restrict__ Wq,
                                 int K, long total_vec) {
  long v = (long)blockIdx.x * blockDim.x + threadIdx.x;  // 8-elem vector index
  if (v >= total_vec) return;
  int vpr = K / 8;
  long row = v / vpr;
  int  i0  = (int)(v % vpr) * 8;      // 8 | 128 -> one scale per vector
  int  g   = i0 / QGROUP;

  float r = rs[row * (K / QGROUP) + g];
  float s = (r > 20.f) ? r : log1pf(expf(r));   // softplus
  s += QEPS;
  unsigned short sb = f2bf_rne(s);

  const float4* p = (const float4*)(wl + row * (long)K + i0);
  float4 a = p[0];
  float4 b = p[1];
  float fs[8] = {a.x, a.y, a.z, a.w, b.x, b.y, b.z, b.w};

  union { unsigned short h[8]; uint4 u; } o;
#pragma unroll
  for (int j = 0; j < 8; ++j)
    o.h[j] = (unsigned short)(sb | ((fs[j] < 0.f) ? 0x8000u : 0u));
  *(uint4*)(Wq + row * (long)K + i0) = o.u;
}

// --- Kernel 2: x f32 -> bf16 (once; keeps VALU cvt out of the GEMM) ---------
__global__ void convert_x_kernel(const float* __restrict__ x,
                                 unsigned short* __restrict__ Xq,
                                 long total_vec) {
  long v = (long)blockIdx.x * blockDim.x + threadIdx.x;
  if (v >= total_vec) return;
  const float4* p = (const float4*)(x + v * 8);
  float4 a = p[0];
  float4 b = p[1];
  float fs[8] = {a.x, a.y, a.z, a.w, b.x, b.y, b.z, b.w};
  union { unsigned short h[8]; uint4 u; } o;
#pragma unroll
  for (int j = 0; j < 8; ++j) o.h[j] = f2bf_rne(fs[j]);
  *(uint4*)(Xq + v * 8) = o.u;
}

// --- Kernel 3: tiled bf16 WMMA GEMM with async double-buffered LDS ----------
// Block tile 128x128x32, 8 waves (2x4); wave tile 64x32 = 4x2 WMMA accums.
__global__ __launch_bounds__(256, 1) void gemm_bf16_kernel(
    const unsigned short* __restrict__ Xq,   // [M][K] bf16 bits
    const unsigned short* __restrict__ Wq,   // [N][K] bf16 bits
    float* __restrict__ C,                   // [M][N]
    int M, int N, int K) {
  __shared__ unsigned short As[2][BM * LDT];
  __shared__ unsigned short Bs[2][BN * LDT];

  const int tid   = threadIdx.x;
  const int lane  = tid & 31;
  const int wid   = tid >> 5;      // 0..7
  const int lrow  = lane & 15;
  const int lhalf = lane >> 4;     // 0/1

  const int wave_m0 = (wid & 1) * 64;   // 2 waves along M
  const int wave_n0 = (wid >> 1) * 32;  // 4 waves along N

  const long block_m0 = (long)blockIdx.y * BM;
  const long block_n0 = (long)blockIdx.x * BN;
  const int  nkt = K / BK;

  // per-thread copy slots: tile is 512 x 16B, 256 threads x 2
  const int v0 = tid,        r0 = v0 >> 2, kc0 = (v0 & 3) * 8;
  const int v1 = tid + 256,  r1 = v1 >> 2, kc1 = (v1 & 3) * 8;
  const unsigned short* gA0 = Xq + (block_m0 + r0) * (long)K + kc0;
  const unsigned short* gA1 = Xq + (block_m0 + r1) * (long)K + kc1;
  const unsigned short* gB0 = Wq + (block_n0 + r0) * (long)K + kc0;
  const unsigned short* gB1 = Wq + (block_n0 + r1) * (long)K + kc1;

  auto issue_tile = [&](int kt, int buf) {
    const long ko = (long)kt * BK;
    async_copy_b128(gA0 + ko, &As[buf][r0 * LDT + kc0]);
    async_copy_b128(gA1 + ko, &As[buf][r1 * LDT + kc1]);
    async_copy_b128(gB0 + ko, &Bs[buf][r0 * LDT + kc0]);
    async_copy_b128(gB1 + ko, &Bs[buf][r1 * LDT + kc1]);
  };

  v8f acc[4][2];
#pragma unroll
  for (int f = 0; f < 4; ++f)
#pragma unroll
    for (int g = 0; g < 2; ++g) acc[f][g] = v8f{};

  issue_tile(0, 0);
  wait_async0();
  __syncthreads();

  for (int kt = 0; kt < nkt; ++kt) {
    const int buf = kt & 1;
    if (kt + 1 < nkt) issue_tile(kt + 1, buf ^ 1);  // DMA overlaps compute

    // B fragment (contiguous K order, not the ISA table order): lane n = column n (= W row),
    // lanes 0-15 K=0..15, lanes 16-31 K=16..31 -> 32 contiguous LDS bytes.
    union { uint4 u[2]; v16bf v; } bfrag[2], afrag[4];
#pragma unroll
    for (int g = 0; g < 2; ++g) {
      const unsigned short* bp =
          &Bs[buf][(wave_n0 + g * 16 + lrow) * LDT + lhalf * 16];
      bfrag[g].u[0] = *(const uint4*)(bp);
      bfrag[g].u[1] = *(const uint4*)(bp + 8);
    }
    // A fragment (same K order as B): lane m = row m, VGPR0-7 hold
    // K = lhalf*16..+15.
#pragma unroll
    for (int f = 0; f < 4; ++f) {
      const unsigned short* ap =
          &As[buf][(wave_m0 + f * 16 + lrow) * LDT + lhalf * 16];
      afrag[f].u[0] = *(const uint4*)(ap);
      afrag[f].u[1] = *(const uint4*)(ap + 8);
    }

#pragma unroll
    for (int f = 0; f < 4; ++f)
#pragma unroll
      for (int g = 0; g < 2; ++g)
        acc[f][g] = __builtin_amdgcn_wmma_f32_16x16x32_bf16(
            false, afrag[f].v, false, bfrag[g].v,
            (short)0, acc[f][g], false, false);

    if (kt + 1 < nkt) {
      wait_async0();     // own async writes of buf^1 landed
      __syncthreads();   // everyone's landed; prior readers of buf^1 done
    }
  }

  // Epilogue: VGPR r holds M=r (lanes 0-15) / M=r+8 (lanes 16-31), N=lane&15.
  // 134 MB streamed out -> non-temporal scalar stores, 64B-coalesced/half-wave.
#pragma unroll
  for (int f = 0; f < 4; ++f) {
    long mbase = block_m0 + wave_m0 + f * 16 + lhalf * 8;
#pragma unroll
    for (int g = 0; g < 2; ++g) {
      long col = block_n0 + wave_n0 + g * 16 + lrow;
#pragma unroll
      for (int r = 0; r < 8; ++r)
        __builtin_nontemporal_store(acc[f][g][r], &C[(mbase + r) * (long)N + col]);
    }
  }
}

// ---------------------------------------------------------------------------
extern "C" void kernel_launch(void* const* d_in, const int* in_sizes, int n_in,
                              void* d_out, int out_size, void* d_ws, size_t ws_size,
                              hipStream_t stream) {
  const float* x  = (const float*)d_in[0];  // [B,S,IN] f32
  const float* wl = (const float*)d_in[1];  // [OUT,IN] f32
  const float* rs = (const float*)d_in[2];  // [OUT,IN/128] f32
  float* out = (float*)d_out;               // [B,S,OUT] f32

  const long xel = (long)in_sizes[0];       // M*K
  const long wel = (long)in_sizes[1];       // N*K
  const long sel = (long)in_sizes[2];       // N*(K/GROUP)
  const long K = (wel / sel) * QGROUP;
  const long N = wel / K;
  const long M = xel / K;

  // workspace: Wq [N*K] bf16 then Xq [M*K] bf16 (72 MB for this shape)
  unsigned short* Wq = (unsigned short*)d_ws;
  unsigned short* Xq = Wq + (size_t)(N * K);

  {
    long wvec = wel / 8;
    dequant_w_kernel<<<dim3((unsigned)((wvec + 255) / 256)), dim3(256), 0, stream>>>(
        wl, rs, Wq, (int)K, wvec);
  }
  {
    long xvec = xel / 8;
    convert_x_kernel<<<dim3((unsigned)((xvec + 255) / 256)), dim3(256), 0, stream>>>(
        x, Xq, xvec);
  }
  {
    dim3 grid((unsigned)(N / BN), (unsigned)(M / BM));
    gemm_bf16_kernel<<<grid, dim3(256), 0, stream>>>(Xq, Wq, out,
                                                     (int)M, (int)N, (int)K);
  }
}
